// SageLayer2_20529943675143
// MI455X (gfx1250) — compile-verified, matrix operand order checked by reading, not run on hardware
//
#include <hip/hip_runtime.h>
#include <hip/hip_bf16.h>
#include <math.h>

typedef __attribute__((ext_vector_type(16))) __bf16 v16bf;
typedef __attribute__((ext_vector_type(8)))  __bf16 v8bf;
typedef __attribute__((ext_vector_type(8)))  float  v8f;

#define D_DIM    256
#define S_CTX    33
#define ROWS_PAD 48
#define LDS_STRIDE 264   // 256 + 8 halves pad -> 528B row stride, bank-conflict free
#define W_ELEMS  (D_DIM * D_DIM)

// ---------------------------------------------------------------------------
// Phase 0: convert Wq|Wk|Wv (f32 row-major [j][k]) to bf16 into workspace.
// ---------------------------------------------------------------------------
__global__ void convert_weights_kernel(const float* __restrict__ Wq,
                                       const float* __restrict__ Wk,
                                       const float* __restrict__ Wv,
                                       __bf16* __restrict__ dst) {
    int i = blockIdx.x * 256 + threadIdx.x;          // 0 .. 3*65536-1
    float v;
    if (i < W_ELEMS)            v = Wq[i];
    else if (i < 2 * W_ELEMS)   v = Wk[i - W_ELEMS];
    else                        v = Wv[i - 2 * W_ELEMS];
    dst[i] = (__bf16)v;
}

// ---------------------------------------------------------------------------
// Fragment loaders. Per-lane layout for bf16 16x16x32 WMMA (ISA 7.12.2):
//   lanes 0-15 : elems 0-7 = K(kc*32+0..7),  elems 8-15 = K(kc*32+16..23)
//   lanes 16-31: elems 0-7 = K(kc*32+8..15), elems 8-15 = K(kc*32+24..31)
// A: row = M index per lane. B: row = N index per lane (W stored row-major,
// B[k][n] = W[n][k], so lane n carries W row n with the same K pattern).
// ---------------------------------------------------------------------------
__device__ __forceinline__ v16bf load_frag(const __bf16* base, int row,
                                           int stride, int kc, int hi) {
    const __bf16* p = base + row * stride + kc * 32 + hi;
    v8bf lo = *(const v8bf*)(p);
    v8bf hh = *(const v8bf*)(p + 16);
    v16bf r;
#pragma unroll
    for (int i = 0; i < 8; ++i) { r[i] = lo[i]; r[i + 8] = hh[i]; }
    return r;
}

__device__ __forceinline__ v8f splat8(float x) {
    v8f v;
#pragma unroll
    for (int i = 0; i < 8; ++i) v[i] = x;
    return v;
}

// ---------------------------------------------------------------------------
// Phase 1: one workgroup (256 thr = 8 waves) per node.
// ---------------------------------------------------------------------------
__global__ void __launch_bounds__(256)
sage_attn_kernel(const int* __restrict__ node_ids,
                 const int* __restrict__ neigh_ids,
                 const float* __restrict__ feat_table,
                 const __bf16* __restrict__ wbf,   // [Wq|Wk|Wv] bf16
                 const float* __restrict__ bq,
                 const float* __restrict__ bk,
                 const float* __restrict__ bv,
                 float* __restrict__ out) {
    __shared__ __align__(16) __bf16 s_ctx[ROWS_PAD * LDS_STRIDE];
    __shared__ float s_scores[ROWS_PAD];
    __shared__ float s_attn[ROWS_PAD];
    __shared__ float s_red;
    __shared__ float s_norm;

    const int tid  = threadIdx.x;
    const int lane = tid & 31;
    const int wv   = tid >> 5;          // wave id 0..7
    const int col  = lane & 15;
    const int hi   = (lane >> 4) * 8;

    const __bf16* wqbf = wbf;
    const __bf16* wkbf = wbf + W_ELEMS;
    const __bf16* wvbf = wbf + 2 * W_ELEMS;

    // ---- Phase A: gather ctx -> bf16 LDS, zero pad rows, zero reducers ----
    const int node = blockIdx.x;
    const int nid  = node_ids[node];
    // row 0 = self
    s_ctx[tid] = (__bf16)feat_table[(size_t)nid * D_DIM + tid];
    // rows 1..32 = neighbors (each iter: all 256 threads cover one row)
#pragma unroll 4
    for (int it = 0; it < 32; ++it) {
        int gid = neigh_ids[node * 32 + it];
        s_ctx[(it + 1) * LDS_STRIDE + tid] =
            (__bf16)feat_table[(size_t)gid * D_DIM + tid];
    }
    // zero rows 33..47 (cols 0..255 only; pad cols never read)
#pragma unroll
    for (int it = 0; it < 15; ++it)
        s_ctx[(S_CTX + it) * LDS_STRIDE + tid] = (__bf16)0.f;
    if (tid < ROWS_PAD) s_scores[tid] = 0.f;
    if (tid == 0) s_norm = 0.f;
    __syncthreads();

    // ---- Phase B: WMMA projections. Wave owns N-tiles {wv, wv+8}.
    //      kc outer / mt inner: each weight fragment feeds all 3 M-tiles,
    //      cutting L2 weight traffic ~2.3x vs. refetch-per-M-tile. ----
    v8f vacc[2][3];
#pragma unroll
    for (int t = 0; t < 2; ++t) {
        const int j0  = (wv + 8 * t) * 16;
        const float bqv = bq[j0 + col];
        const float bkv = bk[j0 + col];
        const float bvv = bv[j0 + col];
        v8f qacc = splat8(bqv);
        v8f kacc[3];
#pragma unroll
        for (int mt = 0; mt < 3; ++mt) {
            kacc[mt]    = splat8(bkv);
            vacc[t][mt] = splat8(bvv);
        }
#pragma unroll
        for (int kc = 0; kc < 8; ++kc) {
            v16bf bK = load_frag(wkbf, j0 + col, D_DIM, kc, hi);
            v16bf bV = load_frag(wvbf, j0 + col, D_DIM, kc, hi);
            v16bf bQ = load_frag(wqbf, j0 + col, D_DIM, kc, hi);
#pragma unroll
            for (int mt = 0; mt < 3; ++mt) {
                v16bf a = load_frag(s_ctx, mt * 16 + col, LDS_STRIDE, kc, hi);
                kacc[mt] = __builtin_amdgcn_wmma_f32_16x16x32_bf16(
                    false, a, false, bK, (short)0, kacc[mt], false, false);
                vacc[t][mt] = __builtin_amdgcn_wmma_f32_16x16x32_bf16(
                    false, a, false, bV, (short)0, vacc[t][mt], false, false);
                if (mt == 0) {
                    qacc = __builtin_amdgcn_wmma_f32_16x16x32_bf16(
                        false, a, false, bQ, (short)0, qacc, false, false);
                }
            }
        }
        // Q row 0 lives in qacc[0], lanes 0..15; broadcast over halves
        const float qv = __shfl(qacc[0], col, 32);
        // partial scores: sum over this wave's 16 columns
#pragma unroll
        for (int mt = 0; mt < 3; ++mt) {
#pragma unroll
            for (int r = 0; r < 8; ++r) {
                float p = qv * kacc[mt][r];
                p += __shfl_xor(p, 1, 32);
                p += __shfl_xor(p, 2, 32);
                p += __shfl_xor(p, 4, 32);
                p += __shfl_xor(p, 8, 32);
                if (lane == 0)        atomicAdd(&s_scores[mt * 16 + r], p);
                else if (lane == 16)  atomicAdd(&s_scores[mt * 16 + r + 8], p);
            }
        }
    }
    __syncthreads();

    // ---- Phase C: softmax over the 33 real scores ----
    if (tid == 0) {
        float mx = s_scores[0];
        for (int k = 1; k < S_CTX; ++k) mx = fmaxf(mx, s_scores[k]);
        s_red = mx;
    }
    __syncthreads();
    if (tid < S_CTX) s_scores[tid] = expf(s_scores[tid] - s_red);
    __syncthreads();
    if (tid == 0) {
        float sm = 0.f;
        for (int k = 0; k < S_CTX; ++k) sm += s_scores[k];
        s_red = 1.0f / sm;
    }
    __syncthreads();
    if (tid < ROWS_PAD) s_attn[tid] = (tid < S_CTX) ? s_scores[tid] * s_red : 0.f;
    __syncthreads();

    // ---- Phase D: mix = attn . V (from resident accumulators), tanh, norm ----
    float tval[2];
#pragma unroll
    for (int t = 0; t < 2; ++t) {
        float mixp = 0.f;
#pragma unroll
        for (int mt = 0; mt < 3; ++mt)
#pragma unroll
            for (int r = 0; r < 8; ++r)
                mixp += s_attn[mt * 16 + r + hi] * vacc[t][mt][r];
        mixp += __shfl_xor(mixp, 16, 32);     // fold half-lane groups
        float tv = tanhf(mixp);
        tval[t] = tv;
        float p2 = tv * tv;
        p2 += __shfl_xor(p2, 1, 32);
        p2 += __shfl_xor(p2, 2, 32);
        p2 += __shfl_xor(p2, 4, 32);
        p2 += __shfl_xor(p2, 8, 32);
        if (lane == 0) atomicAdd(&s_norm, p2);
    }
    __syncthreads();

    const float invn = 1.0f / fmaxf(sqrtf(s_norm), 1e-12f);
    if (lane < 16) {
        out[(size_t)node * D_DIM + wv * 16 + lane]       = tval[0] * invn;
        out[(size_t)node * D_DIM + (wv + 8) * 16 + lane] = tval[1] * invn;
    }
}

// ---------------------------------------------------------------------------
extern "C" void kernel_launch(void* const* d_in, const int* in_sizes, int n_in,
                              void* d_out, int out_size, void* d_ws, size_t ws_size,
                              hipStream_t stream) {
    const int*   node_ids  = (const int*)d_in[0];
    const int*   neigh_ids = (const int*)d_in[1];
    const float* feat      = (const float*)d_in[2];
    const float* Wq        = (const float*)d_in[3];
    const float* bq        = (const float*)d_in[4];
    const float* Wk        = (const float*)d_in[5];
    const float* bk        = (const float*)d_in[6];
    const float* Wv        = (const float*)d_in[7];
    const float* bv        = (const float*)d_in[8];
    float* out = (float*)d_out;
    __bf16* wbf = (__bf16*)d_ws;     // 3 * 65536 * 2B = 384 KB scratch

    const int n_nodes = in_sizes[0];

    convert_weights_kernel<<<(3 * W_ELEMS) / 256, 256, 0, stream>>>(Wq, Wk, Wv, wbf);
    sage_attn_kernel<<<n_nodes, 256, 0, stream>>>(node_ids, neigh_ids, feat, wbf,
                                                  bq, bk, bv, out);
}
